// MaskedMultiHeadAttention_90984587198582
// MI455X (gfx1250) — matrix operand order repaired and checked by reading, not compiled, not run on hardware
//
#include <hip/hip_runtime.h>

// ---------------------------------------------------------------------------
// MaskedMultiHeadAttention forward for MI455X (gfx1250, wave32, WMMA)
//   B=2, S=2048, D=1024, H=16, dk=64
// Pipeline:
//   1) fp32 -> bf16 conversion of x and the 4 weight matrices (RNE)
//   2) bf16 NT-GEMM (x @ W.T + b) -> Q [B,H,S,64], K [B,H,S,64], Vt [B,H,64,S]
//      32x64 output tile per wave: two row-tiles share B-fragments
//   3) flash-attention per (b,h,32-query-tile): online softmax, P@V via LDS
//      transpose; K/V B-fragments shared across the two 16-query A-fragments
//   4) bf16 NT-GEMM (O @ Wo.T + bo) -> fp32 d_out
// All matrix math: v_wmma_f32_16x16x32_bf16.
// ---------------------------------------------------------------------------

typedef __attribute__((ext_vector_type(16))) __bf16         bf16x16;
typedef __attribute__((ext_vector_type(8)))  float          floatx8;
typedef __attribute__((ext_vector_type(8)))  unsigned short ushort8;
typedef __attribute__((ext_vector_type(16))) unsigned short ushort16;

#define SWZ(x, imm) __int_as_float(__builtin_amdgcn_ds_swizzle(__float_as_int(x), (imm)))
// xor-1/2/4/8 within 16-lane halves (group-of-32 encoding, and_mask=0x1f)
#define SWZ_X1 0x041f
#define SWZ_X2 0x081f
#define SWZ_X4 0x101f
#define SWZ_X8 0x201f

__device__ __forceinline__ unsigned short f2bf(float f) {
  unsigned u = __float_as_uint(f);
  u += 0x7fffu + ((u >> 16) & 1u);       // round-to-nearest-even
  return (unsigned short)(u >> 16);
}

__device__ __forceinline__ floatx8 zero8() {
  floatx8 z = {0.f, 0.f, 0.f, 0.f, 0.f, 0.f, 0.f, 0.f};
  return z;
}

// Build a 16x32 bf16 A-fragment from two 8-element (16B) chunks:
// elements 0..7 = K[16h .. 16h+7], elements 8..15 = K[16h+8 .. 16h+15]
__device__ __forceinline__ bf16x16 mk_afrag(ushort8 lo, ushort8 hi) {
  union { ushort16 u; bf16x16 b; } cvt;
  cvt.u = __builtin_shufflevector(lo, hi, 0, 1, 2, 3, 4, 5, 6, 7,
                                          8, 9, 10, 11, 12, 13, 14, 15);
  return cvt.b;
}

__device__ __forceinline__ bf16x16 ld_afrag(const unsigned short* row, int hv) {
  return mk_afrag(*(const ushort8*)(row + 16 * hv),
                  *(const ushort8*)(row + 16 * hv + 8));
}

// One contiguous 32B load = a full per-lane B-fragment slice (16 bf16)
__device__ __forceinline__ bf16x16 ld_bfrag(const unsigned short* p) {
  union { ushort16 u; bf16x16 b; } cvt;
  cvt.u = *(const ushort16*)p;
  return cvt.b;
}

__device__ __forceinline__ floatx8 wmma_bf16(bf16x16 a, bf16x16 b, floatx8 c) {
  return __builtin_amdgcn_wmma_f32_16x16x32_bf16(
      /*neg_a=*/false, a, /*neg_b=*/false, b,
      /*c_mod=*/(short)0, c, /*reuse_a=*/false, /*reuse_b=*/false);
}

// ---------------------------------------------------------------------------
// fp32 -> bf16 conversion
// ---------------------------------------------------------------------------
__global__ void mha_cvt_bf16(const float* __restrict__ in,
                             unsigned short* __restrict__ out, int n) {
  int i = blockIdx.x * blockDim.x + threadIdx.x;
  if (i < n) out[i] = f2bf(in[i]);
}

// ---------------------------------------------------------------------------
// Y[M=4096, N=1024] = A[M,1024] @ W[N,1024]^T + bias   (bf16 in, WMMA)
// One wave per 32x64 output tile (2 row-tiles x 4 col-tiles); B-fragments
// are loaded once per k-step and reused by both row-tiles.
// Epilogue modes:
//   0: bf16 -> [B,H,S,64]   (Q / K)
//   1: bf16 -> [B,H,64,S]   (V transposed)
//   2: fp32 -> row-major [M,N] (final output)
// ---------------------------------------------------------------------------
__global__ __launch_bounds__(32) void mha_gemm_nt(
    const unsigned short* __restrict__ A,
    const unsigned short* __restrict__ W,
    const float* __restrict__ bias,
    unsigned short* __restrict__ out_bf,
    float* __restrict__ out_f32,
    int mode) {
  const int lane = threadIdx.x;
  const int ln = lane & 15;
  const int hv = lane >> 4;
  const int mt = blockIdx.x;   // 0..127 (32-row tile)
  const int ng = blockIdx.y;   // 0..15  (4 column tiles each)

  const unsigned short* arow0 = A + (size_t)(mt * 32 + ln) * 1024;
  const unsigned short* arow1 = A + (size_t)(mt * 32 + 16 + ln) * 1024;

  floatx8 acc[2][4] = {{zero8(), zero8(), zero8(), zero8()},
                       {zero8(), zero8(), zero8(), zero8()}};

  for (int k0 = 0; k0 < 1024; k0 += 32) {
    bf16x16 af[2];
    af[0] = ld_afrag(arow0 + k0, hv);
    af[1] = ld_afrag(arow1 + k0, hv);
#pragma unroll
    for (int j = 0; j < 4; ++j) {
      const int n = (ng * 4 + j) * 16 + ln;
      const unsigned short* wp = W + (size_t)n * 1024 + k0 + hv * 16;
      if (k0 + 32 < 1024) __builtin_prefetch(wp + 32, 0, 1);
      const bf16x16 bf = ld_bfrag(wp);
      acc[0][j] = wmma_bf16(af[0], bf, acc[0][j]);
      acc[1][j] = wmma_bf16(af[1], bf, acc[1][j]);
    }
  }

#pragma unroll
  for (int mi = 0; mi < 2; ++mi) {
#pragma unroll
    for (int j = 0; j < 4; ++j) {
      const int n = (ng * 4 + j) * 16 + ln;
      const float bn = bias[n];
#pragma unroll
      for (int r = 0; r < 8; ++r) {
        const int m = mt * 32 + mi * 16 + r + 8 * hv;
        const float v = acc[mi][j][r] + bn;
        if (mode == 2) {
          out_f32[(size_t)m * 1024 + n] = v;
        } else {
          const int b = m >> 11, s = m & 2047;
          const int h = n >> 6, d = n & 63;
          const size_t idx =
              (mode == 0) ? ((size_t)((b * 16 + h) * 2048 + s) * 64 + d)
                          : ((size_t)((b * 16 + h) * 64 + d) * 2048 + s);
          out_bf[idx] = f2bf(v);
        }
      }
    }
  }
}

// ---------------------------------------------------------------------------
// Flash attention: one wave per (b, h, 32-query tile), 32 keys per step.
// K/V fragments are loaded once per key-block and reused by both 16-query
// A-fragments.  Q,K: [B,H,S,64] bf16; Vt: [B,H,64,S] bf16; O: [B*S,1024] bf16
// ---------------------------------------------------------------------------
__global__ __launch_bounds__(32) void mha_flash_attn(
    const unsigned short* __restrict__ Q,
    const unsigned short* __restrict__ K,
    const unsigned short* __restrict__ Vt,
    unsigned short* __restrict__ O) {
  __shared__ __align__(32) unsigned short pTile[2][16 * 32];  // P as bf16

  const int lane = threadIdx.x;
  const int ln = lane & 15;
  const int hv = lane >> 4;
  const int qt = blockIdx.x;   // 0..63 (32 queries each)
  const int bh = blockIdx.y;   // 0..31
  const int b = bh >> 4, h = bh & 15;

  const unsigned short* Qbh = Q + (size_t)bh * 2048 * 64;
  const unsigned short* Kbh = K + (size_t)bh * 2048 * 64;
  const unsigned short* Vbh = Vt + (size_t)bh * 64 * 2048;

  // Q tiles as 16x32 A-fragments (dk = 64 -> 2 fragments per 16-query tile)
  bf16x16 aq[2][2];
#pragma unroll
  for (int qi = 0; qi < 2; ++qi) {
    const unsigned short* qrow = Qbh + (size_t)(qt * 32 + qi * 16 + ln) * 64;
    aq[qi][0] = ld_afrag(qrow, hv);
    aq[qi][1] = ld_afrag(qrow + 32, hv);
  }

  floatx8 acc[2][4] = {{zero8(), zero8(), zero8(), zero8()},
                       {zero8(), zero8(), zero8(), zero8()}};
  float mrow[2][8], lrow[2][8];
#pragma unroll
  for (int qi = 0; qi < 2; ++qi)
#pragma unroll
    for (int r = 0; r < 8; ++r) { mrow[qi][r] = -1e30f; lrow[qi][r] = 0.f; }

  const float scale = 0.125f;    // 1/sqrt(64)
  const int nkb = qt + 1;        // 32-key blocks touched by causality

  for (int kb = 0; kb < nkb; ++kb) {
    const int kbase = kb * 32;

    // K-tile B-fragments, shared across both query tiles
    bf16x16 bk[2][2];
#pragma unroll
    for (int t = 0; t < 2; ++t) {
      const unsigned short* krow = Kbh + (size_t)(kbase + t * 16 + ln) * 64;
      bk[t][0] = ld_bfrag(krow + hv * 16);
      bk[t][1] = ld_bfrag(krow + 32 + hv * 16);
    }

    // S = (Q @ K^T) * scale, causal mask
    floatx8 sc[2][2];
#pragma unroll
    for (int qi = 0; qi < 2; ++qi) {
#pragma unroll
      for (int t = 0; t < 2; ++t) {
        floatx8 s = zero8();
        s = wmma_bf16(aq[qi][0], bk[t][0], s);
        s = wmma_bf16(aq[qi][1], bk[t][1], s);
        sc[qi][t] = s;
      }
    }
#pragma unroll
    for (int qi = 0; qi < 2; ++qi) {
#pragma unroll
      for (int t = 0; t < 2; ++t) {
        const int kidx = kbase + t * 16 + ln;
#pragma unroll
        for (int r = 0; r < 8; ++r) {
          const int qidx = qt * 32 + qi * 16 + r + 8 * hv;
          const float v = sc[qi][t][r] * scale;
          sc[qi][t][r] = (kidx > qidx) ? -1e9f : v;
        }
      }
    }

    // online softmax update (row = r + 8*hv lives in one 16-lane half)
    float alpha[2][8];
#pragma unroll
    for (int qi = 0; qi < 2; ++qi) {
#pragma unroll
      for (int r = 0; r < 8; ++r) {
        float v = fmaxf(sc[qi][0][r], sc[qi][1][r]);
        v = fmaxf(v, SWZ(v, SWZ_X1));
        v = fmaxf(v, SWZ(v, SWZ_X2));
        v = fmaxf(v, SWZ(v, SWZ_X4));
        v = fmaxf(v, SWZ(v, SWZ_X8));
        const float mnew = fmaxf(mrow[qi][r], v);
        alpha[qi][r] = __expf(mrow[qi][r] - mnew);
        const float p0 = __expf(sc[qi][0][r] - mnew);
        const float p1 = __expf(sc[qi][1][r] - mnew);
        sc[qi][0][r] = p0;
        sc[qi][1][r] = p1;
        float ls = p0 + p1;
        ls += SWZ(ls, SWZ_X1);
        ls += SWZ(ls, SWZ_X2);
        ls += SWZ(ls, SWZ_X4);
        ls += SWZ(ls, SWZ_X8);
        lrow[qi][r] = lrow[qi][r] * alpha[qi][r] + ls;
        mrow[qi][r] = mnew;
      }
#pragma unroll
      for (int j = 0; j < 4; ++j)
#pragma unroll
        for (int r = 0; r < 8; ++r) acc[qi][j][r] *= alpha[qi][r];
    }

    // P (16x32 fp32, C-layout) -> LDS bf16 row-major -> A-layout fragments
    __syncthreads();
#pragma unroll
    for (int qi = 0; qi < 2; ++qi)
#pragma unroll
      for (int t = 0; t < 2; ++t)
#pragma unroll
        for (int r = 0; r < 8; ++r)
          pTile[qi][(r + 8 * hv) * 32 + t * 16 + ln] = f2bf(sc[qi][t][r]);
    __syncthreads();

    bf16x16 ap[2];
#pragma unroll
    for (int qi = 0; qi < 2; ++qi)
      ap[qi] = ld_afrag(&pTile[qi][ln * 32], hv);

    // acc += P @ V   (B from Vt: contiguous along keys; shared across qi)
#pragma unroll
    for (int j = 0; j < 4; ++j) {
      const unsigned short* vr =
          Vbh + (size_t)(j * 16 + ln) * 2048 + kbase + hv * 16;
      const bf16x16 bv = ld_bfrag(vr);
      acc[0][j] = wmma_bf16(ap[0], bv, acc[0][j]);
      acc[1][j] = wmma_bf16(ap[1], bv, acc[1][j]);
    }
  }

  // epilogue: O[b*S + q][h*64 + d] = acc / l
#pragma unroll
  for (int qi = 0; qi < 2; ++qi) {
#pragma unroll
    for (int r = 0; r < 8; ++r) {
      const float inv = 1.f / lrow[qi][r];
      const int m = qt * 32 + qi * 16 + r + 8 * hv;
      const size_t rowoff = ((size_t)(b * 2048 + m)) * 1024 + h * 64;
#pragma unroll
      for (int j = 0; j < 4; ++j)
        O[rowoff + j * 16 + ln] = f2bf(acc[qi][j][r] * inv);
    }
  }
}

// ---------------------------------------------------------------------------
// host-side launcher
// ---------------------------------------------------------------------------
extern "C" void kernel_launch(void* const* d_in, const int* in_sizes, int n_in,
                              void* d_out, int out_size, void* d_ws,
                              size_t ws_size, hipStream_t stream) {
  (void)in_sizes; (void)n_in; (void)out_size; (void)ws_size;
  const float* x  = (const float*)d_in[0];
  // d_in[1] is the causal tril mask; applied analytically in-kernel.
  const float* Wq = (const float*)d_in[2];
  const float* bq = (const float*)d_in[3];
  const float* Wk = (const float*)d_in[4];
  const float* bk = (const float*)d_in[5];
  const float* Wv = (const float*)d_in[6];
  const float* bv = (const float*)d_in[7];
  const float* Wo = (const float*)d_in[8];
  const float* bo = (const float*)d_in[9];
  float* out = (float*)d_out;

  char* ws = (char*)d_ws;
  const size_t MB = (size_t)1 << 20;
  unsigned short* Xbf = (unsigned short*)(ws + 0 * MB);   // 8 MB
  unsigned short* Wqb = (unsigned short*)(ws + 8 * MB);   // 2 MB
  unsigned short* Wkb = (unsigned short*)(ws + 10 * MB);  // 2 MB
  unsigned short* Wvb = (unsigned short*)(ws + 12 * MB);  // 2 MB
  unsigned short* Wob = (unsigned short*)(ws + 14 * MB);  // 2 MB
  unsigned short* Qb  = (unsigned short*)(ws + 16 * MB);  // 8 MB
  unsigned short* Kb  = (unsigned short*)(ws + 24 * MB);  // 8 MB
  unsigned short* Vtb = (unsigned short*)(ws + 32 * MB);  // 8 MB
  unsigned short* Obf = (unsigned short*)(ws + 40 * MB);  // 8 MB

  const int NX = 2 * 2048 * 1024;   // x elements
  const int NW = 1024 * 1024;       // weight elements

  mha_cvt_bf16<<<(NX + 255) / 256, 256, 0, stream>>>(x, Xbf, NX);
  mha_cvt_bf16<<<(NW + 255) / 256, 256, 0, stream>>>(Wq, Wqb, NW);
  mha_cvt_bf16<<<(NW + 255) / 256, 256, 0, stream>>>(Wk, Wkb, NW);
  mha_cvt_bf16<<<(NW + 255) / 256, 256, 0, stream>>>(Wv, Wvb, NW);
  mha_cvt_bf16<<<(NW + 255) / 256, 256, 0, stream>>>(Wo, Wob, NW);

  dim3 gg(128, 16), gb(32, 1, 1);
  mha_gemm_nt<<<gg, gb, 0, stream>>>(Xbf, Wqb, bq, Qb, nullptr, 0);
  mha_gemm_nt<<<gg, gb, 0, stream>>>(Xbf, Wkb, bk, Kb, nullptr, 0);
  mha_gemm_nt<<<gg, gb, 0, stream>>>(Xbf, Wvb, bv, Vtb, nullptr, 1);

  mha_flash_attn<<<dim3(64, 32), gb, 0, stream>>>(Qb, Kb, Vtb, Obf);

  mha_gemm_nt<<<gg, gb, 0, stream>>>(Obf, Wob, bo, nullptr, out, 2);
}
